// MinkFormerBlock_44384192037348
// MI455X (gfx1250) — matrix operand order repaired and checked by reading, not compiled, not run on hardware
//
#include <hip/hip_runtime.h>

#define NROWS 32768
#define CCH   128

typedef __bf16 bf16;
typedef __attribute__((ext_vector_type(16))) __bf16 v16bf;
typedef __attribute__((ext_vector_type(8)))  __bf16 v8bf;
typedef __attribute__((ext_vector_type(8)))  float  v8f;

union AFrag { v16bf v; v8bf h[2]; };

// ---------------------------------------------------------------------------
// Pack fp32 weights [taps][Cin=128][Cout=128] into a contiguous-k WMMA B-fragment
// layout (bf16, 32x16 per fragment): for tap t, k-step s (k0=32s), n-tile j
// (n0=16j), lane l, element e:   value = W[t][k0 + (l>=16?16:0) + e][n0 + l%16]
// Each lane's fragment is then 16 contiguous bf16 (32 bytes).
// ---------------------------------------------------------------------------
__global__ void pack_w_kernel(const float* __restrict__ W, bf16* __restrict__ out, int total) {
  int idx = blockIdx.x * 256 + threadIdx.x;
  if (idx >= total) return;
  int e = idx & 15;
  int l = (idx >> 4) & 31;
  int j = (idx >> 9) & 7;
  int s = (idx >> 12) & 3;
  int t = idx >> 14;
  int K = s * 32 + ((l & 16) ? 16 : 0) + e;
  int n = j * 16 + (l & 15);
  out[idx] = (bf16)W[((size_t)t * CCH + K) * CCH + n];
}

__global__ void cvt_bf16_kernel(const float* __restrict__ in, bf16* __restrict__ out, int n) {
  int i = blockIdx.x * 256 + threadIdx.x;
  if (i < n) out[i] = (bf16)in[i];
}

// ---------------------------------------------------------------------------
// Gather-GEMM-accumulate sparse conv:
//   out[i] = sum_t feats[nbr[i*taps+t]] @ W[t]     (nbr==nullptr -> identity)
// feats: bf16 [N][128]; Wp: packed bf16 B-fragments [taps][4 ksteps][8 ntiles]
// Block: 256 threads = 8 waves; 16 rows/wave, 128 rows/block.
//
// Pipelining (per tap t):
//  - weights for t+1: CDNA5 async DMA to LDS (global_load_async_to_lds_b128,
//    ASYNCcnt), double-buffered in LDS
//  - gather index + A-fragments for t+1: prefetched into a register double
//    buffer while the 32 WMMAs of tap t execute
//  - B-fragments: software-pipelined one fragment ahead so each WMMA has the
//    next ds_load pair already in flight
// mode 0: outF = acc ; mode 1: outB = bf16(acc) ; mode 2: outB = bf16(acc*Vgate)
// ---------------------------------------------------------------------------
__global__ __launch_bounds__(256) void spconv_kernel(
    const bf16* __restrict__ feats, const int* __restrict__ nbr, int taps,
    const bf16* __restrict__ Wp,
    const float* __restrict__ Vgate, float* __restrict__ outF, bf16* __restrict__ outB,
    int mode)
{
  __shared__ alignas(64) bf16 ldsW[2 * 16384];   // 2 x 32KB tap buffers
  const int tid  = threadIdx.x;
  const int lane = tid & 31;
  const int wave = tid >> 5;
  const int rowBase = blockIdx.x * 128 + wave * 16;
  const int m  = lane & 15;
  const bool hi = lane >= 16;

  v8f acc[8] = {};

  // Low 32 bits of the generic __shared__ pointer = workgroup-relative LDS
  // byte offset (flat aperture keeps the LDS offset in addr[31:0]).
  const unsigned ldsBase = (unsigned)(unsigned long long)(const void*)ldsW;

  // Async-copy one 32KB tap into an LDS buffer: 256 threads x 8 async B128.
  // Hardware DMA tracked by ASYNCcnt; no VGPR data, no blocking ds_store.
  auto issueTap = [&](int buf, int t) {
    unsigned long long gbase = (unsigned long long)(const void*)(Wp + (size_t)t * 16384);
    unsigned ldsOff = ldsBase + (unsigned)buf * 32768u + (unsigned)tid * 128u;
    unsigned gOff   = (unsigned)tid * 128u;
    #pragma unroll
    for (int i = 0; i < 8; ++i) {
      asm volatile("global_load_async_to_lds_b128 %0, %1, %2"
                   :: "v"(ldsOff + i * 16u), "v"(gOff + i * 16u), "s"(gbase)
                   : "memory");
    }
  };

  // A fragments, contiguous K layout (matches packed B): lanes 0-15 hold K
  // [k0..k0+15]; lanes 16-31 hold [k0+16..k0+31].
  auto loadA = [&](AFrag* a, int row) {
    const bf16* arow = feats + (size_t)row * CCH;
    #pragma unroll
    for (int s = 0; s < 4; ++s) {
      int k0 = s * 32 + (hi ? 16 : 0);
      a[s].h[0] = *(const v8bf*)(arow + k0);
      a[s].h[1] = *(const v8bf*)(arow + k0 + 8);
    }
  };

  issueTap(0, 0);
  AFrag aCur[4], aNext[4];
  {
    int row0 = nbr ? nbr[(size_t)(rowBase + m) * taps] : (rowBase + m);
    loadA(aCur, row0);
  }
  asm volatile("s_wait_asynccnt 0x0" ::: "memory");
  __syncthreads();

  for (int t = 0; t < taps; ++t) {
    const int buf = t & 1;
    if (t + 1 < taps) {
      issueTap(buf ^ 1, t + 1);                     // DMA weights for t+1
      int rowN = nbr ? nbr[(size_t)(rowBase + m) * taps + (t + 1)] : (rowBase + m);
      loadA(aNext, rowN);                           // prefetch A for t+1
    }

    const bf16* bbase = ldsW + buf * 16384;
    v16bf bcur = *(const v16bf*)(bbase + ((unsigned)lane << 4));
    #pragma unroll
    for (int u = 0; u < 32; ++u) {                  // u = s*8 + j
      v16bf bnext = bcur;
      if (u < 31) bnext = *(const v16bf*)(bbase + ((unsigned)((u + 1) * 32 + lane) << 4));
      acc[u & 7] = __builtin_amdgcn_wmma_f32_16x16x32_bf16(
          false, aCur[u >> 3].v, false, bcur, (short)0, acc[u & 7], false, false);
      bcur = bnext;
    }

    // Own async copies for t+1 must have landed before anyone reads that
    // buffer next iteration; barrier publishes them block-wide.
    asm volatile("s_wait_asynccnt 0x0" ::: "memory");
    __syncthreads();

    if (t + 1 < taps) {
      #pragma unroll
      for (int s = 0; s < 4; ++s) aCur[s] = aNext[s];
    }
  }

  // Epilogue: C/D layout lanes 0-15: N=lane, M=v ; lanes 16-31: N=lane-16, M=8+v
  const int n  = lane & 15;
  const int mb = hi ? 8 : 0;
  #pragma unroll
  for (int j = 0; j < 8; ++j) {
    int col = j * 16 + n;
    #pragma unroll
    for (int v = 0; v < 8; ++v) {
      size_t o = (size_t)(rowBase + mb + v) * CCH + col;
      float val = acc[j][v];
      if (mode == 2)      { val *= Vgate[o]; outB[o] = (bf16)val; }
      else if (mode == 1) { outB[o] = (bf16)val; }
      else                { outF[o] = val; }
    }
  }
}

// ---------------------------------------------------------------------------
// Deterministic two-pass batchnorm stats (population mean/var over rows)
// ---------------------------------------------------------------------------
__global__ void bn_stats_kernel(const float* __restrict__ T, float* __restrict__ part) {
  int c = threadIdx.x;           // 128 threads = channels
  int b = blockIdx.x;            // 64 blocks of 512 rows
  float s = 0.f, q = 0.f;
  int r0 = b * (NROWS / 64);
  for (int r = r0; r < r0 + (NROWS / 64); ++r) {
    float v = T[(size_t)r * CCH + c];
    s += v; q += v * v;
  }
  part[b * CCH + c] = s;
  part[64 * CCH + b * CCH + c] = q;
}

__global__ void bn_finalize_kernel(const float* __restrict__ part,
                                   const float* __restrict__ gamma,
                                   const float* __restrict__ beta,
                                   float* __restrict__ sb) {
  int c = threadIdx.x;
  float s = 0.f, q = 0.f;
  for (int b = 0; b < 64; ++b) { s += part[b * CCH + c]; q += part[64 * CCH + b * CCH + c]; }
  float mu  = s * (1.0f / NROWS);
  float var = q * (1.0f / NROWS) - mu * mu;
  float sc  = rsqrtf(var + 1e-5f) * gamma[c];
  sb[c]       = sc;
  sb[CCH + c] = beta[c] - mu * sc;
}

// out1 = BN1(T)+x (kept fp32 for residual2), R = bf16(relu(out1)) for next conv
__global__ void bn_res_relu_kernel(const float* __restrict__ T, const float* __restrict__ x,
                                   const float* __restrict__ sb,
                                   float* __restrict__ O1, bf16* __restrict__ R, int n) {
  int i = blockIdx.x * 256 + threadIdx.x;
  if (i >= n) return;
  int c = i & (CCH - 1);
  float o = T[i] * sb[c] + sb[CCH + c] + x[i];
  O1[i] = o;
  R[i]  = (bf16)fmaxf(o, 0.f);
}

// final: relu(BN2(T) + out1)
__global__ void bn_res_relu_out_kernel(const float* __restrict__ T, const float* __restrict__ O1,
                                       const float* __restrict__ sb,
                                       float* __restrict__ out, int n) {
  int i = blockIdx.x * 256 + threadIdx.x;
  if (i >= n) return;
  int c = i & (CCH - 1);
  float o = T[i] * sb[c] + sb[CCH + c] + O1[i];
  out[i] = fmaxf(o, 0.f);
}

// ---------------------------------------------------------------------------
extern "C" void kernel_launch(void* const* d_in, const int* in_sizes, int n_in,
                              void* d_out, int out_size, void* d_ws, size_t ws_size,
                              hipStream_t stream)
{
  (void)in_sizes; (void)n_in; (void)out_size; (void)ws_size;
  const float* x   = (const float*)d_in[0];
  const float* Wa1 = (const float*)d_in[1];
  const float* Wv1 = (const float*)d_in[2];
  const float* W5  = (const float*)d_in[3];
  const float* W31 = (const float*)d_in[4];
  const float* W32 = (const float*)d_in[5];
  const float* g1  = (const float*)d_in[6];
  const float* b1  = (const float*)d_in[7];
  const float* g2  = (const float*)d_in[8];
  const float* b2  = (const float*)d_in[9];
  const int* nbr5  = (const int*)d_in[10];
  const int* nbr3a = (const int*)d_in[11];
  const int* nbr3b = (const int*)d_in[12];

  char* ws = (char*)d_ws;
  size_t off = 0;
  auto alloc = [&](size_t bytes) -> char* {
    char* p = ws + off; off += (bytes + 255) & ~(size_t)255; return p;
  };
  const size_t NC = (size_t)NROWS * CCH;
  bf16*  xb   = (bf16*) alloc(NC * 2);
  bf16*  A    = (bf16*) alloc(NC * 2);
  bf16*  P    = (bf16*) alloc(NC * 2);
  bf16*  R    = (bf16*) alloc(NC * 2);
  float* V    = (float*)alloc(NC * 4);
  float* T    = (float*)alloc(NC * 4);          // reused for conv1 & conv2 outputs
  float* O1   = (float*)alloc(NC * 4);
  bf16*  Wa1p = (bf16*) alloc((size_t)16384 * 2);
  bf16*  Wv1p = (bf16*) alloc((size_t)16384 * 2);
  bf16*  W5p  = (bf16*) alloc((size_t)125 * 16384 * 2);
  bf16*  W31p = (bf16*) alloc((size_t)27 * 16384 * 2);
  bf16*  W32p = (bf16*) alloc((size_t)27 * 16384 * 2);
  float* part = (float*)alloc((size_t)2 * 64 * CCH * 4);
  float* sb1  = (float*)alloc(2 * CCH * 4);
  float* sb2  = (float*)alloc(2 * CCH * 4);

  const int nc = (int)NC;
  cvt_bf16_kernel<<<nc / 256, 256, 0, stream>>>(x, xb, nc);
  pack_w_kernel<<<16384 / 256, 256, 0, stream>>>(Wa1, Wa1p, 16384);
  pack_w_kernel<<<16384 / 256, 256, 0, stream>>>(Wv1, Wv1p, 16384);
  pack_w_kernel<<<125 * 16384 / 256, 256, 0, stream>>>(W5, W5p, 125 * 16384);
  pack_w_kernel<<<27 * 16384 / 256, 256, 0, stream>>>(W31, W31p, 27 * 16384);
  pack_w_kernel<<<27 * 16384 / 256, 256, 0, stream>>>(W32, W32p, 27 * 16384);

  dim3 grid(NROWS / 128), blk(256);
  // A = bf16(x @ Wa1)
  spconv_kernel<<<grid, blk, 0, stream>>>(xb, nullptr, 1, Wa1p, nullptr, nullptr, A, 1);
  // V = x @ Wv1 (fp32)
  spconv_kernel<<<grid, blk, 0, stream>>>(xb, nullptr, 1, Wv1p, nullptr, V, nullptr, 0);
  // P = bf16( sparse_conv(A, nbr5, W5) * V )   (gated product, 125 taps)
  spconv_kernel<<<grid, blk, 0, stream>>>(A, nbr5, 125, W5p, V, nullptr, P, 2);
  // T = sparse_conv(P, nbr3a, W31)
  spconv_kernel<<<grid, blk, 0, stream>>>(P, nbr3a, 27, W31p, nullptr, T, nullptr, 0);
  // BN1 + residual(x), R = relu in bf16
  bn_stats_kernel<<<64, 128, 0, stream>>>(T, part);
  bn_finalize_kernel<<<1, 128, 0, stream>>>(part, g1, b1, sb1);
  bn_res_relu_kernel<<<nc / 256, 256, 0, stream>>>(T, x, sb1, O1, R, nc);
  // T = sparse_conv(R, nbr3b, W32)
  spconv_kernel<<<grid, blk, 0, stream>>>(R, nbr3b, 27, W32p, nullptr, T, nullptr, 0);
  // BN2 + residual(O1) + relu -> d_out (fp32)
  bn_stats_kernel<<<64, 128, 0, stream>>>(T, part);
  bn_finalize_kernel<<<1, 128, 0, stream>>>(part, g2, b2, sb2);
  bn_res_relu_out_kernel<<<nc / 256, 256, 0, stream>>>(T, O1, sb2, (float*)d_out, nc);
}
